// LongformerXLLayer_12403865551022
// MI455X (gfx1250) — matrix operand order repaired and checked by reading, compiled, not run on hardware
//
#include <hip/hip_runtime.h>

// ---------------------------------------------------------------------------
// LongformerXL layer for MI455X (gfx1250): bf16 WMMA everywhere, async
// global->LDS double-buffered GEMM pipeline (ASYNCcnt), banded flash-attn.
// B=1, S=2048, M=512, D=1024, F=4096, H=16, DH=64, W=256, L=2560.
// ---------------------------------------------------------------------------

typedef __bf16 bf16_t;
typedef bf16_t v16bf __attribute__((ext_vector_type(16)));
typedef bf16_t v8bf  __attribute__((ext_vector_type(8)));
typedef float  v8f   __attribute__((ext_vector_type(8)));

union FragAB { v16bf v; v8bf h[2]; };

#define S_   2048
#define M_   512
#define D_   1024
#define F_   4096
#define H_   16
#define DH_  64
#define W_   256
#define L_   (M_ + S_)      // 2560
#define T_   (2 * W_ + 1)   // 513 valid relative distances
#define TP_  528            // 513 padded up to multiple of 16

static __device__ __forceinline__ float gelu_tanh(float x) {
  const float c0 = 0.7978845608028654f;
  const float x3 = x * x * x;
  return 0.5f * x * (1.0f + tanhf(c0 * (x + 0.044715f * x3)));
}

// low 32 bits of a flat LDS pointer == LDS byte offset (ISA aperture rule)
static __device__ __forceinline__ unsigned lds_off(const void* p) {
  return (unsigned)(unsigned long long)p;
}

// ---------------------------------------------------------------------------
// Generic bf16 GEMM: C[M,N] = A[M,K] * B^T[N,K] (+bias, +gelu), fp32 accum.
// A row-major (lda), B stored N-major over K (ldb), batched via blockIdx.z.
// 128x128 tile, BK=32, 8 waves, each wave: 2(m) x 4(n) wmma tiles.
// Staging: GLOBAL_LOAD_ASYNC_TO_LDS_B128 double-buffered pipeline; each wave
// issues exactly 4 async ops per stage, so s_wait_asynccnt 0x4 releases the
// current buffer while the next stage is still in flight.
// OOB tile rows/cols are clamped (not zeroed): garbage in A row m' / B col n'
// only reaches C row m' / col n', which the guarded epilogue never stores.
// Kdim must be a multiple of 32 (uses here: 1024, 4096, 64).
// ---------------------------------------------------------------------------
template<bool OUT_BF16, bool HAS_BIAS, bool DO_GELU>
__global__ __launch_bounds__(256)
void gemm_bf16_kernel(const bf16_t* __restrict__ A, int lda, long long strideA,
                      const bf16_t* __restrict__ B, int ldb, long long strideB,
                      void* __restrict__ Cptr, int ldc, long long strideC,
                      const float* __restrict__ bias,
                      int Mdim, int Ndim, int Kdim) {
  __shared__ bf16_t As[2][128][32];   // 16 KB
  __shared__ bf16_t Bs[2][128][32];   // 16 KB

  const int tid  = threadIdx.x;
  const int lane = tid & 31;
  const int wave = tid >> 5;          // 0..7
  const int wm   = wave >> 1;         // 0..3 : 32-row slab
  const int wn   = wave & 1;          // 0..1 : 64-col slab
  const int bm0  = blockIdx.y * 128;
  const int bn0  = blockIdx.x * 128;

  A += (long long)blockIdx.z * strideA;
  B += (long long)blockIdx.z * strideB;

  v8f acc[2][4];
#pragma unroll
  for (int i = 0; i < 2; ++i)
#pragma unroll
    for (int j = 0; j < 4; ++j)
#pragma unroll
      for (int g = 0; g < 8; ++g) acc[i][j][g] = 0.0f;

  // Loader mapping: thread -> (row 0..127, 16-elem K-half 0/16).
  const int lrow = tid >> 1;
  const int lcol = (tid & 1) << 4;
  int grA = bm0 + lrow; if (grA >= Mdim) grA = Mdim - 1;   // clamp, see above
  int gnB = bn0 + lrow; if (gnB >= Ndim) gnB = Ndim - 1;
  const bf16_t* gArow = A + (long long)grA * lda;
  const bf16_t* gBrow = B + (long long)gnB * ldb;

  const unsigned ldsA = lds_off(&As[0][lrow][lcol]);
  const unsigned ldsB = lds_off(&Bs[0][lrow][lcol]);
  const unsigned bufStride = (unsigned)sizeof(bf16_t) * 128 * 32;  // 8192 B

  auto issue_stage = [&](int k0, int buf) {
    const bf16_t* pa = gArow + k0 + lcol;
    const bf16_t* pb = gBrow + k0 + lcol;
    const unsigned la = ldsA + (unsigned)buf * bufStride;
    const unsigned lb = ldsB + (unsigned)buf * bufStride;
    asm volatile("global_load_async_to_lds_b128 %0, %1, off"
                 :: "v"(la), "v"(pa) : "memory");
    asm volatile("global_load_async_to_lds_b128 %0, %1, off offset:16"
                 :: "v"(la), "v"(pa) : "memory");
    asm volatile("global_load_async_to_lds_b128 %0, %1, off"
                 :: "v"(lb), "v"(pb) : "memory");
    asm volatile("global_load_async_to_lds_b128 %0, %1, off offset:16"
                 :: "v"(lb), "v"(pb) : "memory");
  };

  issue_stage(0, 0);

  const int arow = lane & 15;
  const int kh8  = (lane >> 4) << 3;
  const int kh16 = (lane >> 4) << 4;

  int buf = 0;
  for (int k0 = 0; k0 < Kdim; k0 += 32) {
    const bool more = (k0 + 32) < Kdim;
    if (more) {
      issue_stage(k0 + 32, buf ^ 1);                 // prefetch next stage
      asm volatile("s_wait_asynccnt 0x4" ::: "memory");  // current stage landed
    } else {
      asm volatile("s_wait_asynccnt 0x0" ::: "memory");
    }
    __syncthreads();   // all waves' async writes for `buf` visible

    // Fragment loads per documented wave32 VGPR layouts:
    //  A (16x32 bf16): lane half kh: elems = K[kh*8..+7] then K[16+kh*8..+7]
    //  B (32x16 bf16): lane n = col, half kh: elems = K[kh*8..+7] then K[16+kh*8..+7]
    FragAB af[2], bfr[4];
#pragma unroll
    for (int i = 0; i < 2; ++i) {
      const int r = wm * 32 + i * 16 + arow;
      af[i].h[0] = *(const v8bf*)&As[buf][r][kh8];
      af[i].h[1] = *(const v8bf*)&As[buf][r][16 + kh8];
    }
#pragma unroll
    for (int j = 0; j < 4; ++j) {
      const int n = wn * 64 + j * 16 + arow;
      bfr[j].h[0] = *(const v8bf*)&Bs[buf][n][kh8];
      bfr[j].h[1] = *(const v8bf*)&Bs[buf][n][16 + kh8];
    }
#pragma unroll
    for (int i = 0; i < 2; ++i)
#pragma unroll
      for (int j = 0; j < 4; ++j)
        acc[i][j] = __builtin_amdgcn_wmma_f32_16x16x32_bf16(
            false, af[i].v, false, bfr[j].v, (short)0, acc[i][j], false, false);
    __syncthreads();   // all waves done reading `buf` before it is refilled
    buf ^= 1;
  }

  // Epilogue. C tile layout: VGPR g -> row m = g + 8*(lane>>4), col n = lane&15.
  const int khc  = lane >> 4;
  const int ncol = lane & 15;
#pragma unroll
  for (int j = 0; j < 4; ++j) {
    const int n = bn0 + wn * 64 + j * 16 + ncol;
    if (n >= Ndim) continue;
    const float bv = HAS_BIAS ? bias[n] : 0.0f;
#pragma unroll
    for (int i = 0; i < 2; ++i) {
      const int mb = bm0 + wm * 32 + i * 16 + khc * 8;
#pragma unroll
      for (int g = 0; g < 8; ++g) {
        const int m = mb + g;
        if (m >= Mdim) continue;
        float val = acc[i][j][g] + bv;
        if (DO_GELU) val = gelu_tanh(val);
        const long long idx =
            (long long)blockIdx.z * strideC + (long long)m * ldc + n;
        if (OUT_BF16) ((bf16_t*)Cptr)[idx] = (bf16_t)val;
        else          ((float*)Cptr)[idx]  = val;
      }
    }
  }
}

// ---------------------------------------------------------------------------
// Banded flash-attention. One wave per (16-query tile, head).
// 32 keys / iter: 4 wmma for AC (qu . k^T), BD gathered from precomputed
// BDall (bf16), online softmax (shfl_xor row reductions), P relayout via LDS,
// 4 wmma for P.V against pre-transposed vT[D][L].
// ---------------------------------------------------------------------------
__global__ __launch_bounds__(32)
void attn_band_kernel(const bf16_t* __restrict__ qu,
                      const bf16_t* __restrict__ kk,
                      const bf16_t* __restrict__ vT,
                      const bf16_t* __restrict__ BD,
                      bf16_t* __restrict__ o) {
  const int qt   = blockIdx.x;        // 0..127
  const int h    = blockIdx.y;        // 0..15
  const int lane = threadIdx.x;       // 0..31
  const int qi0  = qt << 4;
  const int row  = lane & 15;
  const int kh   = lane >> 4;

  __shared__ bf16_t Pl[16][32];

  FragAB qa[2];  // qu A-fragments for the 2 K-chunks of DH=64
#pragma unroll
  for (int c = 0; c < 2; ++c) {
    const bf16_t* base = qu + (long long)(qi0 + row) * D_ + h * DH_ + c * 32;
    qa[c].h[0] = *(const v8bf*)(base + kh * 8);
    qa[c].h[1] = *(const v8bf*)(base + 16 + kh * 8);
  }

  v8f acc[4];
  float Mrow[8], Srow[8];
#pragma unroll
  for (int t = 0; t < 4; ++t)
#pragma unroll
    for (int g = 0; g < 8; ++g) acc[t][g] = 0.0f;
#pragma unroll
  for (int g = 0; g < 8; ++g) { Mrow[g] = -1e30f; Srow[g] = 0.0f; }

  int jstart = (M_ + qi0 - W_) & ~31;           // >= 0 always (M > W)
  if (jstart < 0) jstart = 0;
  int jend = M_ + qi0 + 15 + W_;
  if (jend > L_ - 1) jend = L_ - 1;

  for (int jb = jstart; jb <= jend; jb += 32) {
    v8f s[2];
#pragma unroll
    for (int nt = 0; nt < 2; ++nt)
#pragma unroll
      for (int g = 0; g < 8; ++g) s[nt][g] = 0.0f;

    // ---- AC = (q+u) . k^T
#pragma unroll
    for (int nt = 0; nt < 2; ++nt) {
      const int j = jb + nt * 16 + row;                 // B lane col = key
      const bf16_t* kbase =
          kk + (long long)(j < L_ ? j : (L_ - 1)) * D_ + h * DH_;
#pragma unroll
      for (int c = 0; c < 2; ++c) {
        FragAB kb;
        kb.h[0] = *(const v8bf*)(kbase + c * 32 + kh * 8);
        kb.h[1] = *(const v8bf*)(kbase + c * 32 + 16 + kh * 8);
        s[nt] = __builtin_amdgcn_wmma_f32_16x16x32_bf16(
            false, qa[c].v, false, kb.v, (short)0, s[nt], false, false);
      }
    }

    // ---- scale + BD gather + band mask
#pragma unroll
    for (int nt = 0; nt < 2; ++nt)
#pragma unroll
      for (int g = 0; g < 8; ++g) {
        const int m = g + (kh << 3);
        const int q = qi0 + m;
        const int j = jb + nt * 16 + row;
        const int t = M_ + q - j + W_;                  // dist + W
        float val = -1e30f;
        if (j < L_ && t >= 0 && t <= 2 * W_) {
          const float bd = (float)BD[((long long)h * S_ + q) * TP_ + t];
          val = (s[nt][g] + bd) * 0.125f;               // DH^-0.5
        }
        s[nt][g] = val;
      }

    // ---- online softmax (rows striped over 16-lane halves)
    float al[8];
#pragma unroll
    for (int g = 0; g < 8; ++g) {
      float tm = fmaxf(s[0][g], s[1][g]);
      tm = fmaxf(tm, __shfl_xor(tm, 1, 32));
      tm = fmaxf(tm, __shfl_xor(tm, 2, 32));
      tm = fmaxf(tm, __shfl_xor(tm, 4, 32));
      tm = fmaxf(tm, __shfl_xor(tm, 8, 32));
      const float nm = fmaxf(Mrow[g], tm);
      al[g] = __expf(Mrow[g] - nm);
      Mrow[g] = nm;
    }
#pragma unroll
    for (int nt = 0; nt < 2; ++nt)
#pragma unroll
      for (int g = 0; g < 8; ++g)
        s[nt][g] = (s[nt][g] > -1e29f) ? __expf(s[nt][g] - Mrow[g]) : 0.0f;
#pragma unroll
    for (int g = 0; g < 8; ++g) {
      float su = s[0][g] + s[1][g];
      su += __shfl_xor(su, 1, 32);
      su += __shfl_xor(su, 2, 32);
      su += __shfl_xor(su, 4, 32);
      su += __shfl_xor(su, 8, 32);
      Srow[g] = Srow[g] * al[g] + su;
    }
#pragma unroll
    for (int t = 0; t < 4; ++t)
#pragma unroll
      for (int g = 0; g < 8; ++g) acc[t][g] *= al[g];

    // ---- P: C-layout -> A-layout via LDS bounce (single wave; LDS in-order)
#pragma unroll
    for (int nt = 0; nt < 2; ++nt)
#pragma unroll
      for (int g = 0; g < 8; ++g)
        Pl[g + (kh << 3)][nt * 16 + row] = (bf16_t)s[nt][g];
    asm volatile("s_wait_dscnt 0" ::: "memory");

    FragAB pf;
    pf.h[0] = *(const v8bf*)&Pl[row][kh * 8];
    pf.h[1] = *(const v8bf*)&Pl[row][16 + kh * 8];

    // ---- P . V  (vT is [D][L]: contiguous over keys)
#pragma unroll
    for (int nt = 0; nt < 4; ++nt) {
      FragAB vb;
      const bf16_t* vbase =
          vT + (long long)(h * DH_ + nt * 16 + row) * L_ + jb + kh * 8;
      vb.h[0] = *(const v8bf*)(vbase);
      vb.h[1] = *(const v8bf*)(vbase + 16);
      acc[nt] = __builtin_amdgcn_wmma_f32_16x16x32_bf16(
          false, pf.v, false, vb.v, (short)0, acc[nt], false, false);
    }
  }

  // ---- finalize: o[q][h*64+dh] = acc / rowsum
#pragma unroll
  for (int nt = 0; nt < 4; ++nt)
#pragma unroll
    for (int g = 0; g < 8; ++g) {
      const int m = g + (kh << 3);
      const float inv = 1.0f / Srow[g];
      o[(long long)(qi0 + m) * D_ + h * DH_ + nt * 16 + row] =
          (bf16_t)(acc[nt][g] * inv);
    }
}

// ---------------------------------------------------------------------------
// Residual add + LayerNorm over D=1024 (one block per row).
// ---------------------------------------------------------------------------
template<bool WRITE_BF16>
__global__ __launch_bounds__(256)
void add_ln_kernel(const float* __restrict__ a, const float* __restrict__ b,
                   const float* __restrict__ gam, const float* __restrict__ bet,
                   float* __restrict__ of, bf16_t* __restrict__ ob) {
  const int r   = blockIdx.x;
  const int tid = threadIdx.x;
  const float* pa = a + (long long)r * D_;
  const float* pb = b + (long long)r * D_;
  __shared__ float red0[8], red1[8];
  float s = 0.0f, s2 = 0.0f;
  for (int i = tid; i < D_; i += 256) {
    const float v = pa[i] + pb[i];
    s += v; s2 += v * v;
  }
  for (int off = 16; off > 0; off >>= 1) {
    s  += __shfl_xor(s,  off, 32);
    s2 += __shfl_xor(s2, off, 32);
  }
  const int lane = tid & 31, wv = tid >> 5;
  if (lane == 0) { red0[wv] = s; red1[wv] = s2; }
  __syncthreads();
  s = 0.0f; s2 = 0.0f;
#pragma unroll
  for (int i = 0; i < 8; ++i) { s += red0[i]; s2 += red1[i]; }
  const float mu  = s  * (1.0f / D_);
  const float var = s2 * (1.0f / D_) - mu * mu;
  const float rs  = rsqrtf(var + 1e-5f);
  for (int i = tid; i < D_; i += 256) {
    const float v = pa[i] + pb[i];
    const float y = (v - mu) * rs * gam[i] + bet[i];
    of[(long long)r * D_ + i] = y;
    if (WRITE_BF16) ob[(long long)r * D_ + i] = (bf16_t)y;
  }
}

// ---------------------------------------------------------------------------
// Small elementwise helpers.
// ---------------------------------------------------------------------------
__global__ void cvt_f32_bf16_kernel(const float* __restrict__ src,
                                    bf16_t* __restrict__ dst, long long n) {
  const long long i = (long long)blockIdx.x * 256 + threadIdx.x;
  if (i < n) dst[i] = (bf16_t)src[i];
}

// W[K][N] fp32 -> WT[N][K] bf16 (tiled through LDS).
__global__ __launch_bounds__(256)
void transpose_cvt_kernel(const float* __restrict__ src,
                          bf16_t* __restrict__ dst, int K, int N) {
  __shared__ float tile[32][33];
  const int k0 = blockIdx.y * 32;
  const int n0 = blockIdx.x * 32;
  const int tx = threadIdx.x & 31;
  const int ty = threadIdx.x >> 5;   // 0..7
  for (int i = ty; i < 32; i += 8) {
    const int k = k0 + i, n = n0 + tx;
    tile[i][tx] = (k < K && n < N) ? src[(long long)k * N + n] : 0.0f;
  }
  __syncthreads();
  for (int i = ty; i < 32; i += 8) {
    const int n = n0 + i, k = k0 + tx;
    if (n < N && k < K) dst[(long long)n * K + k] = (bf16_t)tile[tx][i];
  }
}

// Sinusoid embedding for relative distances dist = t - W, t in [0, 513).
__global__ void relemb_kernel(bf16_t* __restrict__ emb) {
  const long long i = (long long)blockIdx.x * 256 + threadIdx.x;
  if (i >= (long long)TP_ * D_) return;
  const int t = (int)(i >> 10);
  const int d = (int)(i & (D_ - 1));
  float val = 0.0f;
  if (t < T_) {
    const float dist = (float)(t - W_);
    const int jj  = (d < D_ / 2) ? d : d - D_ / 2;
    const float inv = __powf(10000.0f, -(2.0f * (float)jj) / (float)D_);
    const float ang = dist * inv;
    val = (d < D_ / 2) ? __sinf(ang) : __cosf(ang);
  }
  emb[i] = (bf16_t)val;
}

// qu = bf16(q + u), qv = bf16(q + v)   (u, v broadcast over [H][DH] == [D])
__global__ void q_bias_kernel(const float* __restrict__ q,
                              const float* __restrict__ u,
                              const float* __restrict__ v,
                              bf16_t* __restrict__ qu,
                              bf16_t* __restrict__ qv) {
  const long long i = (long long)blockIdx.x * 256 + threadIdx.x;
  if (i >= (long long)S_ * D_) return;
  const int d = (int)(i & (D_ - 1));
  const float ql = q[i];
  qu[i] = (bf16_t)(ql + u[d]);
  qv[i] = (bf16_t)(ql + v[d]);
}

// ---------------------------------------------------------------------------
// Launch
// ---------------------------------------------------------------------------
extern "C" void kernel_launch(void* const* d_in, const int* in_sizes, int n_in,
                              void* d_out, int out_size, void* d_ws,
                              size_t ws_size, hipStream_t stream) {
  const float* x    = (const float*)d_in[0];
  const float* mem  = (const float*)d_in[1];
  const float* Wq   = (const float*)d_in[2];
  const float* Wk   = (const float*)d_in[3];
  const float* Wv   = (const float*)d_in[4];
  const float* Wo   = (const float*)d_in[5];
  const float* Wr   = (const float*)d_in[6];
  const float* u    = (const float*)d_in[7];
  const float* vpb  = (const float*)d_in[8];
  const float* ln1g = (const float*)d_in[9];
  const float* ln1b = (const float*)d_in[10];
  const float* W1   = (const float*)d_in[11];
  const float* b1   = (const float*)d_in[12];
  const float* W2   = (const float*)d_in[13];
  const float* b2   = (const float*)d_in[14];
  const float* ln2g = (const float*)d_in[15];
  const float* ln2b = (const float*)d_in[16];

  char* ws = (char*)d_ws;
  size_t off = 0;
  auto carve = [&](size_t bytes) -> void* {
    void* p = ws + off;
    off += (bytes + 255) & ~(size_t)255;
    return p;
  };

  bf16_t* x_bf  = (bf16_t*)carve((size_t)S_ * D_ * 2);
  bf16_t* kv_bf = (bf16_t*)carve((size_t)L_ * D_ * 2);
  bf16_t* WqT   = (bf16_t*)carve((size_t)D_ * D_ * 2);
  bf16_t* WkT   = (bf16_t*)carve((size_t)D_ * D_ * 2);
  bf16_t* WvT   = (bf16_t*)carve((size_t)D_ * D_ * 2);
  bf16_t* WoT   = (bf16_t*)carve((size_t)D_ * D_ * 2);
  bf16_t* WrT   = (bf16_t*)carve((size_t)D_ * D_ * 2);
  bf16_t* W1T   = (bf16_t*)carve((size_t)D_ * F_ * 2);
  bf16_t* W2T   = (bf16_t*)carve((size_t)F_ * D_ * 2);
  bf16_t* Remb  = (bf16_t*)carve((size_t)TP_ * D_ * 2);
  bf16_t* R_bf  = (bf16_t*)carve((size_t)TP_ * D_ * 2);
  float*  q_f32 = (float*) carve((size_t)S_ * D_ * 4);
  bf16_t* qu_bf = (bf16_t*)carve((size_t)S_ * D_ * 2);
  bf16_t* qv_bf = (bf16_t*)carve((size_t)S_ * D_ * 2);
  bf16_t* k_bf  = (bf16_t*)carve((size_t)L_ * D_ * 2);
  bf16_t* vT_bf = (bf16_t*)carve((size_t)D_ * L_ * 2 + 256);  // pad: PV overread
  bf16_t* BD_bf = (bf16_t*)carve((size_t)H_ * S_ * TP_ * 2);
  bf16_t* o_bf  = (bf16_t*)carve((size_t)S_ * D_ * 2);
  float*  aout  = (float*) carve((size_t)S_ * D_ * 4);
  float*  h_f32 = (float*) carve((size_t)S_ * D_ * 4);
  bf16_t* h_bf  = (bf16_t*)carve((size_t)S_ * D_ * 2);
  bf16_t* g_bf  = (bf16_t*)carve((size_t)S_ * F_ * 2);
  float*  f_f32 = (float*) carve((size_t)S_ * D_ * 4);

  const dim3 blk256(256);
  const long long nXD = (long long)S_ * D_, nMD = (long long)M_ * D_;

  // 1. bf16 conversions: x, kv = [mem; x]
  cvt_f32_bf16_kernel<<<(unsigned)((nXD + 255) / 256), blk256, 0, stream>>>(x, x_bf, nXD);
  cvt_f32_bf16_kernel<<<(unsigned)((nMD + 255) / 256), blk256, 0, stream>>>(mem, kv_bf, nMD);
  cvt_f32_bf16_kernel<<<(unsigned)((nXD + 255) / 256), blk256, 0, stream>>>(x, kv_bf + nMD, nXD);

  // 2. transpose weights to [N][K] bf16
  transpose_cvt_kernel<<<dim3(D_ / 32, D_ / 32), blk256, 0, stream>>>(Wq, WqT, D_, D_);
  transpose_cvt_kernel<<<dim3(D_ / 32, D_ / 32), blk256, 0, stream>>>(Wk, WkT, D_, D_);
  transpose_cvt_kernel<<<dim3(D_ / 32, D_ / 32), blk256, 0, stream>>>(Wv, WvT, D_, D_);
  transpose_cvt_kernel<<<dim3(D_ / 32, D_ / 32), blk256, 0, stream>>>(Wo, WoT, D_, D_);
  transpose_cvt_kernel<<<dim3(D_ / 32, D_ / 32), blk256, 0, stream>>>(Wr, WrT, D_, D_);
  transpose_cvt_kernel<<<dim3(F_ / 32, D_ / 32), blk256, 0, stream>>>(W1, W1T, D_, F_);
  transpose_cvt_kernel<<<dim3(D_ / 32, F_ / 32), blk256, 0, stream>>>(W2, W2T, F_, D_);

  // 3. relative sinusoid embedding (banded distances only)
  relemb_kernel<<<(unsigned)(((long long)TP_ * D_ + 255) / 256), blk256, 0, stream>>>(Remb);

  // 4. q = x @ Wq (fp32)
  gemm_bf16_kernel<false, false, false><<<dim3(D_ / 128, S_ / 128, 1), blk256, 0, stream>>>(
      x_bf, D_, 0, WqT, D_, 0, q_f32, D_, 0, nullptr, S_, D_, D_);
  // 5. qu/qv = bf16(q + u/v)
  q_bias_kernel<<<(unsigned)((nXD + 255) / 256), blk256, 0, stream>>>(q_f32, u, vpb, qu_bf, qv_bf);
  // 6. k = kv @ Wk (bf16)
  gemm_bf16_kernel<true, false, false><<<dim3(D_ / 128, L_ / 128, 1), blk256, 0, stream>>>(
      kv_bf, D_, 0, WkT, D_, 0, k_bf, D_, 0, nullptr, L_, D_, D_);
  // 7. vT[D][L] = (kv @ Wv)^T  via  C = WvT * kv^T
  gemm_bf16_kernel<true, false, false><<<dim3(L_ / 128, D_ / 128, 1), blk256, 0, stream>>>(
      WvT, D_, 0, kv_bf, D_, 0, vT_bf, L_, 0, nullptr, D_, L_, D_);
  // 8. R = Remb @ Wr (bf16)
  gemm_bf16_kernel<true, false, false><<<dim3(D_ / 128, (TP_ + 127) / 128, 1), blk256, 0, stream>>>(
      Remb, D_, 0, WrT, D_, 0, R_bf, D_, 0, nullptr, TP_, D_, D_);
  // 9. BDall[h][q][t] = (q+v)_h . R_h[t]  — batched over heads (grid.z)
  gemm_bf16_kernel<true, false, false><<<dim3((TP_ + 127) / 128, S_ / 128, H_), blk256, 0, stream>>>(
      qv_bf, D_, (long long)DH_, R_bf, D_, (long long)DH_,
      BD_bf, TP_, (long long)S_ * TP_, nullptr, S_, TP_, DH_);
  // 10. banded flash-attention
  attn_band_kernel<<<dim3(S_ / 16, H_), dim3(32), 0, stream>>>(qu_bf, k_bf, vT_bf, BD_bf, o_bf);
  // 11. attention output projection
  gemm_bf16_kernel<false, false, false><<<dim3(D_ / 128, S_ / 128, 1), blk256, 0, stream>>>(
      o_bf, D_, 0, WoT, D_, 0, aout, D_, 0, nullptr, S_, D_, D_);
  // 12. h = LN1(x + o)
  add_ln_kernel<true><<<S_, blk256, 0, stream>>>(x, aout, ln1g, ln1b, h_f32, h_bf);
  // 13. FFN1: gelu(h @ W1 + b1) -> bf16
  gemm_bf16_kernel<true, true, true><<<dim3(F_ / 128, S_ / 128, 1), blk256, 0, stream>>>(
      h_bf, D_, 0, W1T, D_, 0, g_bf, F_, 0, b1, S_, F_, D_);
  // 14. FFN2: g @ W2 + b2 -> fp32
  gemm_bf16_kernel<false, true, false><<<dim3(D_ / 128, S_ / 128, 1), blk256, 0, stream>>>(
      g_bf, F_, 0, W2T, F_, 0, f_f32, D_, 0, b2, S_, D_, F_);
  // 15. out = LN2(h + f)
  add_ln_kernel<false><<<S_, blk256, 0, stream>>>(h_f32, f_f32, ln2g, ln2b,
                                                  (float*)d_out, nullptr);
}
